// EdgeFrame_29248727286342
// MI455X (gfx1250) — compile-verified, matrix operand order checked by reading, not run on hardware
//
#include <hip/hip_runtime.h>
#include <hip/hip_bf16.h>
#include <math.h>

typedef _Float16 v8h  __attribute__((ext_vector_type(8)));
typedef _Float16 v16h __attribute__((ext_vector_type(16)));
typedef float    v8f  __attribute__((ext_vector_type(8)));

#define TILE 256          // edges per block
#define APAD 72           // f16 row pitch (halves) for A/B tiles (bank-conflict free B128)
#define HPAD 68           // f32 row pitch for h2 tile (16B-aligned rows, low conflicts)
#define EPS  1e-5f

// Load one 16-half WMMA fragment (A or B operand) from an LDS row, per the
// CDNA5 16-bit A layout: halves 0..7 = K(koff..koff+7), halves 8..15 = K(16+koff..).
__device__ __forceinline__ v16h load_frag(const _Float16* rowp, int koff) {
    v8h lo = *(const v8h*)(rowp + koff);
    v8h hi = *(const v8h*)(rowp + 16 + koff);
    v16h r;
#pragma unroll
    for (int i = 0; i < 8; ++i) { r[i] = lo[i]; r[i + 8] = hi[i]; }
    return r;
}

__global__ __launch_bounds__(256) void edgeframe_kernel(
    const float* __restrict__ pos, const float* __restrict__ vel,
    const long long* __restrict__ ei,
    const float* __restrict__ W1, const float* __restrict__ b1,
    const float* __restrict__ g1, const float* __restrict__ be1,
    const float* __restrict__ W2, const float* __restrict__ b2,
    const float* __restrict__ g2, const float* __restrict__ be2,
    int E, float* __restrict__ out)
{
    __shared__ __align__(16) _Float16 sA[TILE * APAD];     // layer-1 output, f16 (A tiles)
    __shared__ __align__(16) _Float16 sW2T[64 * APAD];     // W2 transposed, f16 (B tiles)
    __shared__ __align__(16) float    sH2[TILE * HPAD];    // layer-2 pre-LN, f32
    __shared__ __align__(16) float    sB2[64], sG2[64], sBe2[64];
    __shared__ float sMu[TILE], sRs[TILE];

    const int tid = threadIdx.x;

    // ---- Phase 0: stage W2^T (f16) + layer-2 vectors into LDS -------------
#pragma unroll
    for (int i = tid; i < 64 * 64; i += 256) {
        int k = i >> 6, n = i & 63;
        sW2T[n * APAD + k] = (_Float16)W2[i];
    }
    if (tid < 64) { sB2[tid] = b2[tid]; sG2[tid] = g2[tid]; sBe2[tid] = be2[tid]; }
    __syncthreads();

    // ---- Phase 1: per-thread edge features -> 8x64 GEMM -> LN -> ReLU -----
    {
        int e = blockIdx.x * TILE + tid;
        int ec = e < E ? e : (E - 1);
        int si = (int)ei[ec];
        int ti = (int)ei[(long long)E + ec];

        float px = pos[ti * 3 + 0] - pos[si * 3 + 0];
        float py = pos[ti * 3 + 1] - pos[si * 3 + 1];
        float pz = pos[ti * 3 + 2] - pos[si * 3 + 2];
        float rn = sqrtf(px * px + py * py + pz * pz);
        float vx = vel[ti * 3 + 0] - vel[si * 3 + 0];
        float vy = vel[ti * 3 + 1] - vel[si * 3 + 1];
        float vz = vel[ti * 3 + 2] - vel[si * 3 + 2];
        float vn = sqrtf(vx * vx + vy * vy + vz * vz);
        float ft[8] = {px, py, pz, rn, vx, vy, vz, vn};

        float h[64];
        const float4* W1v = (const float4*)W1;   // (8,64) row-major
#pragma unroll
        for (int n0 = 0; n0 < 64; n0 += 4) {
            float4 acc = *(const float4*)&b1[n0];          // uniform -> s_load
#pragma unroll
            for (int k = 0; k < 8; ++k) {
                float4 w = W1v[k * 16 + (n0 >> 2)];        // uniform -> s_load
                acc.x += ft[k] * w.x; acc.y += ft[k] * w.y;
                acc.z += ft[k] * w.z; acc.w += ft[k] * w.w;
            }
            h[n0] = acc.x; h[n0 + 1] = acc.y; h[n0 + 2] = acc.z; h[n0 + 3] = acc.w;
        }
        float s = 0.f;
#pragma unroll
        for (int n = 0; n < 64; ++n) s += h[n];
        float mu = s * (1.0f / 64.0f);
        float s2 = 0.f;
#pragma unroll
        for (int n = 0; n < 64; ++n) { float d = h[n] - mu; s2 += d * d; }
        float rs = rsqrtf(s2 * (1.0f / 64.0f) + EPS);

        _Float16* arow = &sA[tid * APAD];
#pragma unroll
        for (int n0 = 0; n0 < 64; n0 += 8) {
            v8h v;
#pragma unroll
            for (int q = 0; q < 8; ++q) {
                int n = n0 + q;
                float val = (h[n] - mu) * rs * g1[n] + be1[n];
                v[q] = (_Float16)fmaxf(val, 0.0f);
            }
            *(v8h*)(arow + n0) = v;
        }
    }
    __syncthreads();

    // ---- Phase 2: layer 2 via WMMA: (16x64) x (64x64) per M-tile ----------
    {
        const int lane = tid & 31;
        const int wv   = tid >> 5;
        const int lrow = lane & 15;
        const int koff = (lane >> 4) * 8;
        const int rhi  = (lane >> 4) * 8;

#pragma unroll
        for (int t = 0; t < 2; ++t) {
            const int mt = wv * 2 + t;
            const _Float16* ar = &sA[(mt * 16 + lrow) * APAD];
            v16h a0 = load_frag(ar, koff);        // K 0..31
            v16h a1 = load_frag(ar + 32, koff);   // K 32..63
#pragma unroll
            for (int nt = 0; nt < 4; ++nt) {
                const _Float16* br = &sW2T[(nt * 16 + lrow) * APAD];
                v16h b0 = load_frag(br, koff);
                v16h b1f = load_frag(br + 32, koff);
                v8f c = {};
                c = __builtin_amdgcn_wmma_f32_16x16x32_f16(false, a0, false, b0,  (short)0, c, false, false);
                c = __builtin_amdgcn_wmma_f32_16x16x32_f16(false, a1, false, b1f, (short)0, c, false, false);
                const int col = nt * 16 + lrow;
                const float bb = sB2[col];
                const int rbase = mt * 16 + rhi;
#pragma unroll
                for (int r = 0; r < 8; ++r)
                    sH2[(rbase + r) * HPAD + col] = c[r] + bb;
            }
        }
    }
    __syncthreads();

    // ---- Phase 3: per-row LN2 statistics ----------------------------------
    {
        const float* row = &sH2[tid * HPAD];
        float s = 0.f, s2 = 0.f;
#pragma unroll
        for (int j = 0; j < 64; j += 4) {
            float4 v = *(const float4*)&row[j];
            s  += v.x + v.y + v.z + v.w;
            s2 += v.x * v.x + v.y * v.y + v.z * v.z + v.w * v.w;
        }
        float mu = s * (1.0f / 64.0f);
        float var = s2 * (1.0f / 64.0f) - mu * mu;
        sMu[tid] = mu;
        sRs[tid] = rsqrtf(var + EPS);
    }
    __syncthreads();

    // ---- Phase 4: coalesced LN2+ReLU store (float4) -----------------------
    {
        const long long base = (long long)blockIdx.x * TILE;
#pragma unroll
        for (int it = 0; it < 16; ++it) {
            int f4  = it * 256 + tid;           // float4 index within tile
            int row = f4 >> 4;
            int j   = (f4 & 15) * 4;
            long long e = base + row;
            if (e < E) {
                float mu = sMu[row], rs = sRs[row];
                float4 v = *(const float4*)&sH2[row * HPAD + j];
                float4 gg = *(const float4*)&sG2[j];
                float4 bb = *(const float4*)&sBe2[j];
                v.x = fmaxf((v.x - mu) * rs * gg.x + bb.x, 0.0f);
                v.y = fmaxf((v.y - mu) * rs * gg.y + bb.y, 0.0f);
                v.z = fmaxf((v.z - mu) * rs * gg.z + bb.z, 0.0f);
                v.w = fmaxf((v.w - mu) * rs * gg.w + bb.w, 0.0f);
                *(float4*)&out[e * 64 + j] = v;
            }
        }
    }
}

extern "C" void kernel_launch(void* const* d_in, const int* in_sizes, int n_in,
                              void* d_out, int out_size, void* d_ws, size_t ws_size,
                              hipStream_t stream) {
    const float*     pos = (const float*)d_in[0];
    const float*     vel = (const float*)d_in[1];
    const long long* ei  = (const long long*)d_in[2];
    const float* W1  = (const float*)d_in[3];
    const float* b1  = (const float*)d_in[4];
    const float* g1  = (const float*)d_in[5];
    const float* be1 = (const float*)d_in[6];
    const float* W2  = (const float*)d_in[7];
    const float* b2  = (const float*)d_in[8];
    const float* g2  = (const float*)d_in[9];
    const float* be2 = (const float*)d_in[10];

    int E = in_sizes[2] / 2;
    int blocks = (E + TILE - 1) / TILE;
    edgeframe_kernel<<<blocks, 256, 0, stream>>>(pos, vel, ei, W1, b1, g1, be1,
                                                 W2, b2, g2, be2, E, (float*)d_out);
}
